// TransformerEncoder_11768210391723
// MI455X (gfx1250) — matrix operand order repaired and checked by reading, not compiled, not run on hardware
//
#include <hip/hip_runtime.h>
#include <hip/hip_bf16.h>
#include <math.h>

// ---------------------------------------------------------------------------
// Types / WMMA helpers (CDNA5 gfx1250, wave32)
// ---------------------------------------------------------------------------
typedef __bf16 bf16_t;
typedef __attribute__((ext_vector_type(16))) __bf16       v16bf;
typedef __attribute__((ext_vector_type(8)))  float        v8f;
typedef __attribute__((ext_vector_type(4)))  unsigned int u32x4;
typedef __attribute__((ext_vector_type(8)))  int          i32x8;
typedef __attribute__((ext_vector_type(4)))  int          i32x4;

#define WMMA_BF16(a, b, c)                                                     \
  __builtin_amdgcn_wmma_f32_16x16x32_bf16(false, (a), false, (b), (short)0,    \
                                          (c), false, false)

#if defined(__has_builtin)
#if __has_builtin(__builtin_amdgcn_tensor_load_to_lds) && \
    __has_builtin(__builtin_amdgcn_s_wait_tensorcnt)
#define HAVE_TDM 1
#endif
#endif
#ifndef HAVE_TDM
#define HAVE_TDM 0
#endif

#if __has_include(<hip/amd_detail/amd_gfx1250_TDM.h>)
#define TDM_SIX_ARG 1   // amdgpu-toolchain (clang-23): 6-arg builtin
#else
#define TDM_SIX_ARG 0   // ROCm 7.2 (clang-22): 5-arg builtin
#endif

union FragAB { v16bf v; u32x4 q[2]; };
union V32    { u32x4 q[4]; bf16_t e[32]; };

// A-fragment (16x32 MxK bf16): lane L (L<16) holds row M=L, K in {0..7,16..23};
// lane L+16 holds row M=L, K in {8..15,24..31}  (ISA 7.12.2).
__device__ __forceinline__ v16bf load_frag_a(const bf16_t* tile, int ld, int lane) {
  const int half = (lane >> 4) & 1;
  const bf16_t* p = tile + (size_t)(lane & 15) * ld + half * 8;
  FragAB f;
  f.q[0] = *(const u32x4*)(p);
  f.q[1] = *(const u32x4*)(p + 16);
  return f.v;
}

// B-fragment (32x16 KxN bf16): lane L holds column N=L&15;
// lanes 0-15: K in {0..7,16..23}, lanes 16-31: K in {8..15,24..31}.
__device__ __forceinline__ v16bf load_frag_b(const bf16_t* tile, int ld, int lane) {
  const int half = (lane >> 4) & 1;
  const bf16_t* p = tile + (size_t)(lane & 15) * ld + half * 8;
  FragAB f;
  f.q[0] = *(const u32x4*)(p);
  f.q[1] = *(const u32x4*)(p + 16);
  return f.v;
}

#if HAVE_TDM
// ---------------------------------------------------------------------------
// TDM: DMA a 2D bf16 tile [128 rows x 32 cols] (row stride = tensor_k elems)
// from global memory into LDS.  D# packed per CDNA5 ISA ch.8:
//  group0: count=1 | lds_addr | global_addr[56:0] | type=2
//  group1: data_size=1(2B), pad_enable, pad_interval=16 DWORDs (=64B = one
//          row), pad_amount=4 DWORDs (16B) -> LDS row stride 80B = 40 elems,
//          tile_dim0=32, tile_dim1=128, tensor_dim0_stride=tensor_k.
// Issued by one wave; completion tracked on TENSORcnt.
// ---------------------------------------------------------------------------
__device__ __forceinline__ void tdm_load_tile(unsigned lds_off, const bf16_t* gptr,
                                              int tensor_k, int rows_total) {
  unsigned long long ga = (unsigned long long)(uintptr_t)gptr;
  u32x4 g0;
  g0[0] = 1u;                                       // count=1, user mode
  g0[1] = lds_off;                                  // lds_addr (bytes)
  g0[2] = (unsigned)(ga & 0xFFFFFFFFu);             // global_addr[31:0]
  g0[3] = (unsigned)((ga >> 32) & 0x1FFFFFFu) | 0x80000000u;  // [56:32] | type=2
  i32x8 g1;
  g1[0] = (1 << 16)    // data_size = 2 bytes
        | (1 << 20)    // pad_enable
        | (3 << 22)    // pad_interval: 16 DWORDs (64B) between pads
        | (3 << 25);   // pad_amount:   4 DWORDs (16B)
  g1[1] = (int)((tensor_k & 0xFFFF) << 16);                         // tdim0 lo16
  g1[2] = (int)(((tensor_k >> 16) & 0xFFFF) | ((rows_total & 0xFFFF) << 16));
  g1[3] = (int)(((rows_total >> 16) & 0xFFFF) | (32 << 16));        // tile_dim0=32
  g1[4] = 128;                                                      // tile_dim1
  g1[5] = tensor_k;                                                 // stride0 lo32
  g1[6] = 0;
  g1[7] = 0;
  i32x4 z4 = {0, 0, 0, 0};
#if TDM_SIX_ARG
  i32x8 z8 = {0, 0, 0, 0, 0, 0, 0, 0};
  __builtin_amdgcn_tensor_load_to_lds(g0, g1, z4, z4, z8, 0);
#else
  __builtin_amdgcn_tensor_load_to_lds(g0, g1, z4, z4, 0);
#endif
}
#endif  // HAVE_TDM

// ---------------------------------------------------------------------------
// f32 -> bf16 conversion (weights, once per launch)
// ---------------------------------------------------------------------------
__global__ void cvt_f32_bf16_kernel(const float* __restrict__ s,
                                    bf16_t* __restrict__ d, int n) {
  int i = blockIdx.x * blockDim.x + threadIdx.x;
  if (i < n) d[i] = (bf16_t)s[i];
}

// ---------------------------------------------------------------------------
// LayerNorm (f32 in) -> bf16 out.  One block per token, 256 threads, D=768.
// ---------------------------------------------------------------------------
__global__ __launch_bounds__(256)
void ln_bf16_kernel(const float* __restrict__ x, const float* __restrict__ g,
                    const float* __restrict__ bta, bf16_t* __restrict__ y) {
  const int tok = blockIdx.x;
  const int t = threadIdx.x;
  const float* xr = x + (size_t)tok * 768;
  float v0 = xr[t], v1 = xr[t + 256], v2 = xr[t + 512];
  float s = v0 + v1 + v2;
  float ss = v0 * v0 + v1 * v1 + v2 * v2;
#pragma unroll
  for (int off = 16; off; off >>= 1) {
    s  += __shfl_xor(s, off, 32);
    ss += __shfl_xor(ss, off, 32);
  }
  __shared__ float shs[8], shq[8];
  if ((t & 31) == 0) { shs[t >> 5] = s; shq[t >> 5] = ss; }
  __syncthreads();
  s = 0.f; ss = 0.f;
#pragma unroll
  for (int i = 0; i < 8; ++i) { s += shs[i]; ss += shq[i]; }
  const float mu   = s * (1.0f / 768.0f);
  const float var  = ss * (1.0f / 768.0f) - mu * mu;
  const float rstd = rsqrtf(var + 1e-5f);
  bf16_t* yr = y + (size_t)tok * 768;
  yr[t]       = (bf16_t)((v0 - mu) * rstd * g[t]       + bta[t]);
  yr[t + 256] = (bf16_t)((v1 - mu) * rstd * g[t + 256] + bta[t + 256]);
  yr[t + 512] = (bf16_t)((v2 - mu) * rstd * g[t + 512] + bta[t + 512]);
}

// ---------------------------------------------------------------------------
// Tiled GEMM:  C[M,N] = epilogue( A[M,K](bf16) @ W[N,K]^T(bf16) + bias )
// Block tile 128x128, K-step 32, 256 threads = 8 waves (4x2), wave tile 32x64.
// Global->LDS via Tensor Data Mover, double-buffered: TDM for tile kt+1 runs
// while waves execute WMMAs on tile kt (TENSORcnt pipelining).
// MODE 0: f32 out = acc + bias + residual;  MODE 1: bf16 out = acc + bias;
// MODE 2: bf16 out = gelu(acc + bias)   (exact erf GELU)
// ---------------------------------------------------------------------------
template <int MODE>
__global__ __launch_bounds__(256)
void gemm_bf16_kernel(const bf16_t* __restrict__ A, const bf16_t* __restrict__ W,
                      const float* __restrict__ bias, const float* __restrict__ res,
                      float* __restrict__ outf, bf16_t* __restrict__ outb,
                      int M, int N, int K) {
  constexpr int LDT = 40;  // 32 + 8 bf16 pad (TDM pad_amount gives this stride)
  __shared__ __align__(16) bf16_t As[2][128 * LDT];
  __shared__ __align__(16) bf16_t Bs[2][128 * LDT];

  const int t    = threadIdx.x;
  const int lane = t & 31;
  const int wid  = t >> 5;
  const int half = (lane >> 4) & 1;
  const int ln   = lane & 15;
  const int wm0  = (wid & 3) * 32;
  const int wn0  = (wid >> 2) * 64;

  const int blockN = blockIdx.x * 128;
  const int blockM = blockIdx.y * 128;

  const bf16_t* Abase = A + (size_t)blockM * K;
  const bf16_t* Bbase = W + (size_t)blockN * K;

  v8f acc[2][4] = {};
  const int ktiles = K >> 5;

#if HAVE_TDM
  const bool issuer = (wid == 0);
  if (issuer) {  // prime the pipeline: tile 0 -> stage 0
    tdm_load_tile((unsigned)(uintptr_t)&As[0][0], Abase, K, 128);
    tdm_load_tile((unsigned)(uintptr_t)&Bs[0][0], Bbase, K, 128);
  }
  for (int kt = 0; kt < ktiles; ++kt) {
    const int st = kt & 1;
    if (issuer) __builtin_amdgcn_s_wait_tensorcnt(0);  // tile kt landed in LDS
    __syncthreads();  // publish LDS to all waves; stage st^1 readers are done
    if (issuer && (kt + 1 < ktiles)) {  // DMA next tile while we compute
      tdm_load_tile((unsigned)(uintptr_t)&As[st ^ 1][0], Abase + (kt + 1) * 32, K, 128);
      tdm_load_tile((unsigned)(uintptr_t)&Bs[st ^ 1][0], Bbase + (kt + 1) * 32, K, 128);
    }
    v16bf a0 = load_frag_a(&As[st][(wm0 + 0)  * LDT], LDT, lane);
    v16bf a1 = load_frag_a(&As[st][(wm0 + 16) * LDT], LDT, lane);
#pragma unroll
    for (int ni = 0; ni < 4; ++ni) {
      v16bf bfr = load_frag_b(&Bs[st][(wn0 + ni * 16) * LDT], LDT, lane);
      acc[0][ni] = WMMA_BF16(a0, bfr, acc[0][ni]);
      acc[1][ni] = WMMA_BF16(a1, bfr, acc[1][ni]);
    }
  }
#else
  // Fallback: vectorized copy, single-buffered.
  const int lr = t >> 1;
  const int lc = (t & 1) * 16;
  for (int kt = 0; kt < ktiles; ++kt) {
    __syncthreads();
    {
      const u32x4* aq = (const u32x4*)(Abase + (size_t)lr * K + kt * 32 + lc);
      const u32x4* bq = (const u32x4*)(Bbase + (size_t)lr * K + kt * 32 + lc);
      u32x4 a0 = aq[0], a1 = aq[1], b0 = bq[0], b1 = bq[1];
      *(u32x4*)&As[0][lr * LDT + lc + 0] = a0;
      *(u32x4*)&As[0][lr * LDT + lc + 8] = a1;
      *(u32x4*)&Bs[0][lr * LDT + lc + 0] = b0;
      *(u32x4*)&Bs[0][lr * LDT + lc + 8] = b1;
    }
    __syncthreads();
    v16bf a0 = load_frag_a(&As[0][(wm0 + 0)  * LDT], LDT, lane);
    v16bf a1 = load_frag_a(&As[0][(wm0 + 16) * LDT], LDT, lane);
#pragma unroll
    for (int ni = 0; ni < 4; ++ni) {
      v16bf bfr = load_frag_b(&Bs[0][(wn0 + ni * 16) * LDT], LDT, lane);
      acc[0][ni] = WMMA_BF16(a0, bfr, acc[0][ni]);
      acc[1][ni] = WMMA_BF16(a1, bfr, acc[1][ni]);
    }
  }
#endif

  // Epilogue.  C layout: VGPR r -> M = r + 8*half; N = lane&15 (ISA 7.12.2).
#pragma unroll
  for (int ni = 0; ni < 4; ++ni) {
    const int col = blockN + wn0 + ni * 16 + ln;
    const float bn = bias[col];
#pragma unroll
    for (int mi = 0; mi < 2; ++mi) {
#pragma unroll
      for (int r = 0; r < 8; ++r) {
        const int row = blockM + wm0 + mi * 16 + r + half * 8;
        const size_t idx = (size_t)row * N + col;
        float v = acc[mi][ni][r] + bn;
        if (MODE == 0) {
          outf[idx] = v + res[idx];
        } else if (MODE == 1) {
          outb[idx] = (bf16_t)v;
        } else {
          float ge = 0.5f * v * (1.0f + erff(v * 0.70710678118654752f));
          outb[idx] = (bf16_t)ge;
        }
      }
    }
  }
}

// ---------------------------------------------------------------------------
// Flash attention.  qkv: bf16 [B*S, 2304] (q|k|v each 12 heads x 64).
// Grid: B*H*(S/64) blocks, 128 threads = 4 waves; wave w owns 16 query rows.
// Online softmax; the 805 MB score tensor never touches HBM.
// scale = 1/sqrt(768) per the reference.
// ---------------------------------------------------------------------------
__global__ __launch_bounds__(128)
void flash_attn_kernel(const bf16_t* __restrict__ qkv, bf16_t* __restrict__ out) {
  constexpr int LDT = 72;
  __shared__ __align__(16) bf16_t Kt[64 * LDT];       // K tile  [n][k]
  __shared__ __align__(16) bf16_t Vt[64 * LDT];       // V tile transposed [d][n]
  __shared__ __align__(16) bf16_t Pt[4 * 16 * LDT];   // per-wave P tiles

  const int t    = threadIdx.x;
  const int lane = t & 31;
  const int w    = t >> 5;
  const int half = (lane >> 4) & 1;
  const int ln   = lane & 15;

  const int bid = blockIdx.x;
  const int qt  = bid & 15;
  const int h   = (bid >> 4) % 12;
  const int b   = bid / 192;

  const size_t qrow0 = (size_t)(b * 1024 + qt * 64 + w * 16);

  v16bf qa[2];
#pragma unroll
  for (int kk = 0; kk < 64; kk += 32)
    qa[kk >> 5] = load_frag_a(qkv + qrow0 * 2304 + h * 64 + kk, 2304, lane);

  v8f oacc[4] = {};
  float m_r[8], l_r[8];
#pragma unroll
  for (int r = 0; r < 8; ++r) { m_r[r] = -3.0e38f; l_r[r] = 0.0f; }

  const float sc = 0.03608439182435161f;  // 1/sqrt(768)

  for (int j = 0; j < 16; ++j) {
    __syncthreads();
    {
      const int n  = t >> 1;
      const int c0 = (t & 1) * 32;
      const size_t krow = (size_t)(b * 1024 + j * 64 + n) * 2304;
      const u32x4* kq = (const u32x4*)(qkv + krow + 768 + h * 64 + c0);
      u32x4 k0 = kq[0], k1 = kq[1], k2 = kq[2], k3 = kq[3];
      *(u32x4*)&Kt[n * LDT + c0 + 0]  = k0;
      *(u32x4*)&Kt[n * LDT + c0 + 8]  = k1;
      *(u32x4*)&Kt[n * LDT + c0 + 16] = k2;
      *(u32x4*)&Kt[n * LDT + c0 + 24] = k3;
      V32 vu;
      const u32x4* vq = (const u32x4*)(qkv + krow + 1536 + h * 64 + c0);
      vu.q[0] = vq[0]; vu.q[1] = vq[1]; vu.q[2] = vq[2]; vu.q[3] = vq[3];
#pragma unroll
      for (int i = 0; i < 32; ++i)
        Vt[(c0 + i) * LDT + n] = vu.e[i];
      if (j + 1 < 16) {  // prefetch next kv tile -> global_prefetch_b8
        __builtin_prefetch(qkv + krow + (size_t)64 * 2304 + 768 + h * 64 + c0, 0, 1);
        __builtin_prefetch(qkv + krow + (size_t)64 * 2304 + 1536 + h * 64 + c0, 0, 1);
      }
    }
    __syncthreads();

    v8f sAcc[4] = {};
#pragma unroll
    for (int kk = 0; kk < 64; kk += 32) {
      v16bf aq = qa[kk >> 5];
#pragma unroll
      for (int f = 0; f < 4; ++f) {
        v16bf kb = load_frag_b(&Kt[(f * 16) * LDT + kk], LDT, lane);
        sAcc[f] = WMMA_BF16(aq, kb, sAcc[f]);
      }
    }

#pragma unroll
    for (int r = 0; r < 8; ++r) {
      float t4[4];
      float mx = -3.0e38f;
#pragma unroll
      for (int f = 0; f < 4; ++f) { t4[f] = sAcc[f][r] * sc; mx = fmaxf(mx, t4[f]); }
#pragma unroll
      for (int off = 8; off; off >>= 1) mx = fmaxf(mx, __shfl_xor(mx, off, 32));
      const float mnew  = fmaxf(m_r[r], mx);
      const float alpha = __expf(m_r[r] - mnew);
      float lsum = 0.f;
#pragma unroll
      for (int f = 0; f < 4; ++f) {
        float p = __expf(t4[f] - mnew);
        lsum += p;
        Pt[(w * 16 + r + half * 8) * LDT + f * 16 + ln] = (bf16_t)p;
      }
#pragma unroll
      for (int off = 8; off; off >>= 1) lsum += __shfl_xor(lsum, off, 32);
      l_r[r] = l_r[r] * alpha + lsum;
      m_r[r] = mnew;
#pragma unroll
      for (int f = 0; f < 4; ++f) oacc[f][r] *= alpha;
    }

#pragma unroll
    for (int kk = 0; kk < 64; kk += 32) {
      v16bf pa = load_frag_a(&Pt[(w * 16) * LDT + kk], LDT, lane);
#pragma unroll
      for (int f = 0; f < 4; ++f) {
        v16bf vb = load_frag_b(&Vt[(f * 16) * LDT + kk], LDT, lane);
        oacc[f] = WMMA_BF16(pa, vb, oacc[f]);
      }
    }
  }

#pragma unroll
  for (int f = 0; f < 4; ++f) {
#pragma unroll
    for (int r = 0; r < 8; ++r) {
      const float val = oacc[f][r] / l_r[r];
      const size_t row = qrow0 + r + half * 8;
      out[row * 768 + h * 64 + f * 16 + ln] = (bf16_t)val;
    }
  }
}

// ---------------------------------------------------------------------------
// Launch
// ---------------------------------------------------------------------------
extern "C" void kernel_launch(void* const* d_in, const int* in_sizes, int n_in,
                              void* d_out, int out_size, void* d_ws, size_t ws_size,
                              hipStream_t stream) {
  const float* x     = (const float*)d_in[0];
  const float* ln_g  = (const float*)d_in[1];
  const float* ln_b  = (const float*)d_in[2];
  const float* w_qkv = (const float*)d_in[3];
  const float* b_qkv = (const float*)d_in[4];
  const float* w_out = (const float*)d_in[5];
  const float* b_out = (const float*)d_in[6];
  const float* w_fc1 = (const float*)d_in[7];
  const float* b_fc1 = (const float*)d_in[8];
  const float* w_fc2 = (const float*)d_in[9];
  const float* b_fc2 = (const float*)d_in[10];
  float* out = (float*)d_out;

  const int TOK = 16384, D = 768, N3 = 2304, HID = 3072;

  char* ws = (char*)d_ws;
  size_t off = 0;
  auto take = [&](size_t bytes) -> char* {
    char* p = ws + off;
    off = (off + bytes + 255) & ~(size_t)255;
    return p;
  };
  bf16_t* xn   = (bf16_t*)take((size_t)TOK * D * 2);
  bf16_t* qkv  = (bf16_t*)take((size_t)TOK * N3 * 2);
  bf16_t* attn = (bf16_t*)take((size_t)TOK * D * 2);
  float*  x1   = (float*) take((size_t)TOK * D * 4);
  bf16_t* hbuf = (bf16_t*)take((size_t)TOK * HID * 2);
  bf16_t* wqb  = (bf16_t*)take((size_t)N3 * D * 2);
  bf16_t* wob  = (bf16_t*)take((size_t)D * D * 2);
  bf16_t* w1b  = (bf16_t*)take((size_t)HID * D * 2);
  bf16_t* w2b  = (bf16_t*)take((size_t)D * HID * 2);

  auto cvt = [&](const float* s, bf16_t* d, int n) {
    cvt_f32_bf16_kernel<<<(n + 255) / 256, 256, 0, stream>>>(s, d, n);
  };
  cvt(w_qkv, wqb, N3 * D);
  cvt(w_out, wob, D * D);
  cvt(w_fc1, w1b, HID * D);
  cvt(w_fc2, w2b, D * HID);

  ln_bf16_kernel<<<TOK, 256, 0, stream>>>(x, ln_g, ln_b, xn);

  gemm_bf16_kernel<1><<<dim3(N3 / 128, TOK / 128), 256, 0, stream>>>(
      xn, wqb, b_qkv, nullptr, nullptr, qkv, TOK, N3, D);

  flash_attn_kernel<<<16 * 12 * 16, 128, 0, stream>>>(qkv, attn);

  gemm_bf16_kernel<0><<<dim3(D / 128, TOK / 128), 256, 0, stream>>>(
      attn, wob, b_out, x, x1, nullptr, TOK, D, D);

  ln_bf16_kernel<<<TOK, 256, 0, stream>>>(x1, ln_g, ln_b, xn);

  gemm_bf16_kernel<2><<<dim3(HID / 128, TOK / 128), 256, 0, stream>>>(
      xn, w1b, b_fc1, nullptr, nullptr, hbuf, TOK, HID, D);

  gemm_bf16_kernel<0><<<dim3(D / 128, TOK / 128), 256, 0, stream>>>(
      hbuf, w2b, b_fc2, x1, out, nullptr, TOK, D, HID);
}
